// RelationMPNN_3212635537992
// MI455X (gfx1250) — matrix operand order repaired and checked by reading, not compiled, not run on hardware
//
#include <hip/hip_runtime.h>
#include <hip/hip_bf16.h>
#include <math.h>

// ---------------------------------------------------------------------------
// RelationMPNN for MI455X (gfx1250, wave32, WMMA bf16 16x16x32, f32 accum)
//
// Shapes: N=8192, D=H=512, C=4, T=8, E=32768  (T*E = 262144 edge rows)
//
//  * Hoisted gathered-concat GEMMs (Pa/Pb, Qa/Qb) -> node-level GEMMs.
//  * ALL GEMM A operands are bf16, stored PRE-SWIZZLED (bit3<->bit4 of the
//    column index) so each lane's 16 A elements are one contiguous 32B
//    v16bf load -> zero conversion/repack VALU in the WMMA hot loop.
//  * 64x64 wave tile: 16 WMMAs per 32-deep K step, B reused 4x, A reused 4x.
//  * agg einsum as one K=4096 GEMM over a (N, T*H) segment-sum layout.
// ---------------------------------------------------------------------------

typedef __bf16 bf16_t;
typedef __attribute__((ext_vector_type(16))) __bf16 v16bf;
typedef __attribute__((ext_vector_type(8)))  float  v8f;

#define NN 8192
#define DD 512
#define HH 512
#define CCC 4
#define TTT 8
#define EEE 32768

__device__ __forceinline__ bf16_t f2bf(float f) {
  union { float f; unsigned int u; } in; in.f = f;
  unsigned int u = in.u;
  unsigned int r = u + 0x7FFFu + ((u >> 16) & 1u); // round-to-nearest-even
  unsigned short hs = (unsigned short)(r >> 16);
  union { unsigned short us; bf16_t b; } out; out.us = hs;
  return out.b;
}

__device__ __forceinline__ float bf2f(bf16_t b) {
  union { unsigned short us; bf16_t b; } in; in.b = b;
  union { unsigned int u; float f; } out; out.u = ((unsigned int)in.us) << 16;
  return out.f;
}

__device__ __forceinline__ float silu_f(float x) {
  return x / (1.f + expf(-x));
}

// Column swizzle for bf16 A-operand storage: swap bit3 <-> bit4 (self-inverse).
// With this layout, WMMA lane-half hf reads K = {hf*8+0..7, hf*8+16..23} as
// ONE contiguous 16-element (32B) chunk at offset k + hf*16.
__device__ __forceinline__ int swz(int c) {
  return (c & ~24) | ((c & 8) << 1) | ((c & 16) >> 1);
}

// ---------------------------------------------------------------------------
// WMMA GEMM:  Y = act( X[M x K](bf16, swizzled) @ W[Nout x K](bf16)^T
//                      + bias ) + resid
// Block: 256 threads = 8 waves (2x4); block tile 128(M) x 256(N);
// wave tile 64x64 -> acc[4][4], 16 WMMAs per K step.
// Yb output (if any) is written swizzled (so it can feed a later GEMM as A).
// Requires: M % 128 == 0, Nout % 256 == 0, K % 32 == 0 (true for all calls).
// ---------------------------------------------------------------------------
__global__ __launch_bounds__(256)
void gemm_wmma_kernel(const bf16_t* __restrict__ X, int lda,
                      const bf16_t* __restrict__ W, int K,
                      const float* __restrict__ bias,
                      const float* __restrict__ resid,
                      float* __restrict__ Yf,
                      bf16_t* __restrict__ Yb,
                      int ldy, int act) {
  const int lane = threadIdx.x & 31;
  const int wave = threadIdx.x >> 5;
  const int wm   = wave >> 2;          // 0..1
  const int wn   = wave & 3;           // 0..3
  const int hf   = lane >> 4;          // lane half (0/1)
  const int l16  = lane & 15;
  const long mbase = (long)blockIdx.x * 128 + wm * 64;
  const long nbase = (long)blockIdx.y * 256 + wn * 64;

  v8f acc[4][4];
  const v8f vz = {0.f,0.f,0.f,0.f,0.f,0.f,0.f,0.f};
#pragma unroll
  for (int s = 0; s < 4; ++s)
#pragma unroll
    for (int i = 0; i < 4; ++i) acc[s][i] = vz;

  for (int k = 0; k < K; k += 32) {
    // A: pre-swizzled storage -> single contiguous 32B load per subtile.
    v16bf a[4];
#pragma unroll
    for (int s = 0; s < 4; ++s)
      a[s] = *reinterpret_cast<const v16bf*>(
          X + (mbase + s*16 + l16) * (long)lda + k + hf*16);
    // B: lane l16 = W row (output col), elements j<8 -> K = hf*8 + j,
    // j>=8 -> K = 16 + hf*8 + (j-8) (two 8-element runs of a W row).
#pragma unroll
    for (int i = 0; i < 4; ++i) {
      const bf16_t* wp = W + (nbase + i*16 + l16) * (long)K + k + hf*8;
      v16bf b;
#pragma unroll
      for (int j = 0; j < 8; ++j) { b[j] = wp[j]; b[8 + j] = wp[16 + j]; }
#pragma unroll
      for (int s = 0; s < 4; ++s)
        acc[s][i] = __builtin_amdgcn_wmma_f32_16x16x32_bf16(
            false, a[s], false, b, (short)0, acc[s][i], false, false);
    }
  }

  // C/D layout: VGPR r -> row m = hf*8 + r, col n = l16.
#pragma unroll
  for (int s = 0; s < 4; ++s)
#pragma unroll
    for (int i = 0; i < 4; ++i) {
      const int  ncol = (int)nbase + i*16 + l16;
      const float bv = bias ? bias[ncol] : 0.f;
#pragma unroll
      for (int r = 0; r < 8; ++r) {
        const long mrow = mbase + s*16 + hf*8 + r;
        float v = acc[s][i][r] + bv;
        if (act == 1) v = silu_f(v);
        if (resid) v += resid[mrow * (long)ldy + ncol];
        if (Yf) Yf[mrow * (long)ldy + ncol] = v;
        if (Yb) Yb[mrow * (long)ldy + swz(ncol)] = f2bf(v);
      }
    }
}

// ---------------------------------------------------------------------------
// Utility kernels
// ---------------------------------------------------------------------------
__global__ void zero_kernel(float* p, long n) {
  long i = (long)blockIdx.x * 256 + threadIdx.x;
  if (i < n) p[i] = 0.f;
}

// f32 -> bf16 conversion; doSwz=1 applies the A-operand column swizzle.
__global__ void convert_bf16_kernel(const float* __restrict__ src, long srcStride,
                                    bf16_t* __restrict__ dst, long dstStride,
                                    int rows, int cols, int doSwz) {
  long idx = (long)blockIdx.x * 256 + threadIdx.x;
  long total = (long)rows * cols;
  if (idx < total) {
    int r = (int)(idx / cols), c = (int)(idx % cols);
    int cd = doSwz ? swz(c) : c;
    dst[r * dstStride + cd] = f2bf(src[r * srcStride + c]);
  }
}

// cd + raw radial + partial sum of radial^2 over edges (per (c,d), this t)
__global__ void radial_kernel(const float* __restrict__ coord,
                              const int* __restrict__ rowIdx,
                              const int* __restrict__ colIdx,
                              float* __restrict__ cd_out,   // E x 12
                              float* __restrict__ rad_out,  // E x 16
                              float* __restrict__ nrm2,     // 16
                              int E) {
  __shared__ float red[256 * 16];
  int e = blockIdx.x * 256 + threadIdx.x;
  float sq[16];
  if (e < E) {
    int r = rowIdx[e], c = colIdx[e];
    float d[12];
#pragma unroll
    for (int i = 0; i < 12; ++i) d[i] = coord[(long)r*12 + i] - coord[(long)c*12 + i];
#pragma unroll
    for (int i = 0; i < 12; ++i) cd_out[(long)e*12 + i] = d[i];
#pragma unroll
    for (int ci = 0; ci < 4; ++ci)
#pragma unroll
      for (int di = 0; di < 4; ++di) {
        float v = d[ci*3+0]*d[di*3+0] + d[ci*3+1]*d[di*3+1] + d[ci*3+2]*d[di*3+2];
        rad_out[(long)e*16 + ci*4 + di] = v;
        sq[ci*4 + di] = v * v;
      }
  } else {
#pragma unroll
    for (int i = 0; i < 16; ++i) sq[i] = 0.f;
  }
#pragma unroll
  for (int i = 0; i < 16; ++i) red[threadIdx.x * 16 + i] = sq[i];
  __syncthreads();
  for (int s = 128; s > 0; s >>= 1) {
    if ((int)threadIdx.x < s)
      for (int i = 0; i < 16; ++i)
        red[threadIdx.x*16 + i] += red[(threadIdx.x + s)*16 + i];
    __syncthreads();
  }
  if (threadIdx.x < 16) atomicAdd(&nrm2[threadIdx.x], red[threadIdx.x]);
}

__global__ void radial_norm_kernel(float* __restrict__ rad,
                                   const float* __restrict__ nrm2, long n) {
  long idx = (long)blockIdx.x * 256 + threadIdx.x;
  if (idx < n) {
    float nv = sqrtf(nrm2[idx & 15]);
    rad[idx] = rad[idx] / fmaxf(nv, 1e-12f);
  }
}

// u1[e,h] = silu(Pa[row,h] + Pb[col,h] + radial.mW1c + ea*mW1attr + mb1)
// Output written as bf16, column-swizzled (consumed only as WMMA A operand).
__global__ __launch_bounds__(256)
void u1_kernel(const float* __restrict__ Pa, const float* __restrict__ Pb,
               const float* __restrict__ rad, const float* __restrict__ ea,
               const int* __restrict__ rowIdx, const int* __restrict__ colIdx,
               const float* __restrict__ mW1, const float* __restrict__ mb1,
               bf16_t* __restrict__ u1) {
  const int e0 = blockIdx.x * 8;
  const int tid = threadIdx.x;
  __shared__ float srad[8][16];
  __shared__ int   srow[8], scol[8];
  __shared__ float sea[8];
  if (tid < 8) { srow[tid] = rowIdx[e0+tid]; scol[tid] = colIdx[e0+tid]; sea[tid] = ea[e0+tid]; }
  if (tid < 128) srad[tid >> 4][tid & 15] = rad[(long)(e0 + (tid >> 4))*16 + (tid & 15)];
  __syncthreads();
  const int h0 = tid * 2;
  float wc[2][16], wa[2], bb[2];
#pragma unroll
  for (int s = 0; s < 2; ++s) {
    const float* wr = mW1 + (long)(h0 + s) * 1041;
#pragma unroll
    for (int i = 0; i < 16; ++i) wc[s][i] = wr[1024 + i];
    wa[s] = wr[1040]; bb[s] = mb1[h0 + s];
  }
  for (int ee = 0; ee < 8; ++ee) {
    const long r = srow[ee], c = scol[ee];
    const float eav = sea[ee];
#pragma unroll
    for (int s = 0; s < 2; ++s) {
      const int hh = h0 + s;
      float acc = Pa[r*512 + hh] + Pb[c*512 + hh] + bb[s] + eav * wa[s];
#pragma unroll
      for (int i = 0; i < 16; ++i) acc += srad[ee][i] * wc[s][i];
      u1[(long)(e0 + ee)*512 + swz(hh)] = f2bf(silu_f(acc));
    }
  }
}

// seg[row, t*512 + col] += msg[e, col]; msg stored swizzled -> col = swz(pos)
__global__ void segadd_kernel(const bf16_t* __restrict__ msg,
                              const int* __restrict__ rowIdx,
                              float* __restrict__ seg, int t, long n) {
  long idx = (long)blockIdx.x * 256 + threadIdx.x;
  if (idx < n) {
    int e = (int)(idx >> 9), p = (int)(idx & 511);
    atomicAdd(&seg[(long)rowIdx[e]*4096 + t*512 + swz(p)], bf2f(msg[idx]));
  }
}

// w[c] = hid[e,:].cW2[t,c,:]; hid stored swizzled. Scatter cd*w into sumv/cnt.
__global__ __launch_bounds__(256)
void wtrans_kernel(const bf16_t* __restrict__ hid, const float* __restrict__ cW2t,
                   const float* __restrict__ cd, const int* __restrict__ rowIdx,
                   float* __restrict__ sumv, float* __restrict__ cnt) {
  const int e = blockIdx.x;
  const int tid = threadIdx.x;
  __shared__ float red[256][4];
  float p[4] = {0.f, 0.f, 0.f, 0.f};
  for (int hp = tid; hp < 512; hp += 256) {
    float mv = bf2f(hid[(long)e*512 + hp]);   // value of column swz(hp)
    const int hc = swz(hp);
#pragma unroll
    for (int c = 0; c < 4; ++c) p[c] += mv * cW2t[c*512 + hc];
  }
#pragma unroll
  for (int c = 0; c < 4; ++c) red[tid][c] = p[c];
  __syncthreads();
  for (int s = 128; s > 0; s >>= 1) {
    if (tid < s)
      for (int c = 0; c < 4; ++c) red[tid][c] += red[tid + s][c];
    __syncthreads();
  }
  if (tid < 12) {
    int c = tid / 3;
    float tr = cd[(long)e*12 + tid] * red[0][c];
    atomicAdd(&sumv[(long)rowIdx[e]*12 + tid], tr);
  }
  if (tid == 12) atomicAdd(&cnt[rowIdx[e]], 1.0f);
}

__global__ void xcat_kernel(const float* __restrict__ h, const float* __restrict__ agg,
                            bf16_t* __restrict__ xcat, long n) {
  long idx = (long)blockIdx.x * 256 + threadIdx.x;
  if (idx < n) {
    long nrow = idx >> 10; int c = (int)(idx & 1023);
    float v = (c < 512) ? h[nrow*512 + c] : agg[nrow*512 + (c - 512)];
    xcat[nrow*1024 + swz(c)] = f2bf(v);
  }
}

__global__ void xnew_kernel(const float* __restrict__ coord, const float* __restrict__ sumv,
                            const float* __restrict__ cnt, float* __restrict__ out, long n) {
  long idx = (long)blockIdx.x * 256 + threadIdx.x;
  if (idx < n) {
    long nn = idx / 12;
    out[idx] = coord[idx] + sumv[idx] / fmaxf(cnt[nn], 1.0f);
  }
}

// m[te] = sum_h silu(Qa[row,h] + Qb[col,h] + ea*eW1[h,512] + eb1[h]) * eW2[h] + eb2
__global__ __launch_bounds__(256)
void m_kernel(const float* __restrict__ Qa, const float* __restrict__ Qb,
              const float* __restrict__ edge_attr, const int* __restrict__ edge_index,
              const float* __restrict__ eW1, const float* __restrict__ eb1,
              const float* __restrict__ eW2, const float* __restrict__ eb2,
              float* __restrict__ mout) {
  const int te = blockIdx.x;
  const int t = te / EEE, e = te % EEE;
  const long r = edge_index[(long)t*2*EEE + e];
  const long c = edge_index[(long)t*2*EEE + EEE + e];
  const float eav = edge_attr[te];
  const int tid = threadIdx.x;
  float part = 0.f;
  for (int h = tid; h < 512; h += 256) {
    float u = Qa[r*512 + h] + Qb[c*512 + h] + eav * eW1[(long)h*1025 + 512] + eb1[h];
    part += silu_f(u) * eW2[h];
  }
  __shared__ float red[256];
  red[tid] = part;
  __syncthreads();
  for (int s = 128; s > 0; s >>= 1) {
    if (tid < s) red[tid] += red[tid + s];
    __syncthreads();
  }
  if (tid == 0) mout[te] = red[0] + eb2[0];
}

// ---------------------------------------------------------------------------
// Host orchestration
// ---------------------------------------------------------------------------
extern "C" void kernel_launch(void* const* d_in, const int* in_sizes, int n_in,
                              void* d_out, int out_size, void* d_ws, size_t ws_size,
                              hipStream_t stream) {
  const float* h         = (const float*)d_in[0];
  const float* coord     = (const float*)d_in[1];
  const float* edge_attr = (const float*)d_in[2];
  const int*   edge_index= (const int*)  d_in[3];
  const float* mW1 = (const float*)d_in[4];
  const float* mb1 = (const float*)d_in[5];
  const float* mW2 = (const float*)d_in[6];
  const float* mb2 = (const float*)d_in[7];
  const float* nW1 = (const float*)d_in[8];
  const float* nb1 = (const float*)d_in[9];
  const float* nW2 = (const float*)d_in[10];
  const float* nb2 = (const float*)d_in[11];
  const float* eW1 = (const float*)d_in[12];
  const float* eb1 = (const float*)d_in[13];
  const float* eW2 = (const float*)d_in[14];
  const float* eb2 = (const float*)d_in[15];
  const float* relW = (const float*)d_in[16];
  const float* cW1  = (const float*)d_in[17];
  const float* cb1  = (const float*)d_in[18];
  const float* cW2  = (const float*)d_in[19];

  float* out_h = (float*)d_out;                       // N*D
  float* out_x = out_h + (long)NN*DD;                 // N*C*3
  float* out_m = out_x + (long)NN*CCC*3;              // T*E

  // Workspace carve-up (256B aligned slots)
  char* ws = (char*)d_ws;
  size_t off = 0;
  auto carve = [&](size_t bytes) -> char* {
    char* p = ws + off;
    off += (bytes + 255) & ~(size_t)255;
    return p;
  };
  bf16_t* wb_mW1a = (bf16_t*)carve((size_t)512*512*2);
  bf16_t* wb_mW1b = (bf16_t*)carve((size_t)512*512*2);
  bf16_t* wb_mW2  = (bf16_t*)carve((size_t)512*512*2);
  bf16_t* wb_cW1  = (bf16_t*)carve((size_t)TTT*512*512*2);
  bf16_t* wb_relR = (bf16_t*)carve((size_t)512*4096*2);   // (D, T*H)
  bf16_t* wb_nW1  = (bf16_t*)carve((size_t)512*1024*2);
  bf16_t* wb_nW2  = (bf16_t*)carve((size_t)512*512*2);
  bf16_t* wb_eW1a = (bf16_t*)carve((size_t)512*512*2);
  bf16_t* wb_eW1b = (bf16_t*)carve((size_t)512*512*2);
  float*  Pa   = (float*)carve((size_t)NN*512*4);         // reused as Qa
  float*  Pb   = (float*)carve((size_t)NN*512*4);         // reused as Qb
  float*  cdb  = (float*)carve((size_t)TTT*EEE*12*4);
  float*  rad  = (float*)carve((size_t)EEE*16*4);
  float*  nrm2 = (float*)carve((size_t)TTT*16*4);
  bf16_t* u1b  = (bf16_t*)carve((size_t)EEE*512*2);       // reused as hid, tmp1
  bf16_t* msgb = (bf16_t*)carve((size_t)EEE*512*2);       // reused as xcat
  float*  seg  = (float*)carve((size_t)NN*4096*4);        // (N, T*H)
  bf16_t* segb = (bf16_t*)carve((size_t)NN*4096*2);       // swizzled bf16 seg
  bf16_t* hb   = (bf16_t*)carve((size_t)NN*512*2);        // swizzled bf16 h / h_new
  float*  agg  = (float*)carve((size_t)NN*512*4);
  float*  sumv = (float*)carve((size_t)NN*12*4);
  float*  cnt  = (float*)carve((size_t)NN*4);
  (void)ws_size; (void)in_sizes; (void)n_in; (void)out_size;

  auto gemm = [&](const bf16_t* X, int lda, const bf16_t* W, int K,
                  const float* bias, const float* resid,
                  float* Yf, bf16_t* Yb, int ldy, int M, int Nout, int act) {
    dim3 g(M / 128, Nout / 256);
    gemm_wmma_kernel<<<g, 256, 0, stream>>>(X, lda, W, K, bias, resid,
                                            Yf, Yb, ldy, act);
  };
  auto conv = [&](const float* src, long ss, bf16_t* dst, long ds,
                  int rows, int cols, int doSwz) {
    long total = (long)rows * cols;
    convert_bf16_kernel<<<(total + 255) / 256, 256, 0, stream>>>(
        src, ss, dst, ds, rows, cols, doSwz);
  };
  auto zero = [&](float* p, long n) {
    zero_kernel<<<(n + 255) / 256, 256, 0, stream>>>(p, n);
  };

  // ---- Phase 0: weight conversions + activation converts + zeroing ----
  conv(mW1,        1041, wb_mW1a, 512,  512, 512,  0);
  conv(mW1 + 512,  1041, wb_mW1b, 512,  512, 512,  0);
  conv(mW2,        512,  wb_mW2,  512,  512, 512,  0);
  conv(cW1,        512,  wb_cW1,  512,  TTT*512, 512, 0);
  for (int t = 0; t < TTT; ++t)
    conv(relW + (long)t*512*512, 512, wb_relR + (long)t*512, 4096, 512, 512, 0);
  conv(nW1,        1024, wb_nW1,  1024, 512, 1024, 0);
  conv(nW2,        512,  wb_nW2,  512,  512, 512,  0);
  conv(eW1,        1025, wb_eW1a, 512,  512, 512,  0);
  conv(eW1 + 513,  1025, wb_eW1b, 512,  512, 512,  0);
  conv(h,          512,  hb,      512,  NN,  512,  1);   // swizzled A operand

  zero(seg,  (long)NN * 4096);
  zero(sumv, (long)NN * 12);
  zero(cnt,  (long)NN);
  zero(nrm2, (long)TTT * 16);

  // ---- Phase 1: node-level pre-projections (hoisted edge GEMMs) ----
  gemm(hb, 512, wb_mW1a, 512, nullptr, nullptr, Pa, nullptr, 512, NN, 512, 0);
  gemm(hb, 512, wb_mW1b, 512, nullptr, nullptr, Pb, nullptr, 512, NN, 512, 0);

  // ---- Phase 2: per relation type t ----
  for (int t = 0; t < TTT; ++t) {
    const int* rowI = edge_index + (long)t*2*EEE;
    const int* colI = rowI + EEE;
    float* cdt = cdb + (long)t*EEE*12;
    radial_kernel<<<EEE / 256, 256, 0, stream>>>(coord, rowI, colI, cdt, rad,
                                                 nrm2 + t*16, EEE);
    radial_norm_kernel<<<((long)EEE*16 + 255) / 256, 256, 0, stream>>>(
        rad, nrm2 + t*16, (long)EEE*16);
    // u1 = silu(feat @ mW1^T + mb1) assembled from Pa/Pb/radial/attr
    u1_kernel<<<EEE / 8, 256, 0, stream>>>(Pa, Pb, rad, edge_attr + (long)t*EEE,
                                           rowI, colI, mW1, mb1, u1b);
    // msg = silu(u1 @ mW2^T + mb2)  (swizzled bf16 A, swizzled bf16 out)
    gemm(u1b, 512, wb_mW2, 512, mb2, nullptr, nullptr, msgb, 512, EEE, 512, 1);
    // seg[row, t*512+col] += msg
    segadd_kernel<<<((long)EEE*512 + 255) / 256, 256, 0, stream>>>(
        msgb, rowI, seg, t, (long)EEE*512);
    // hid = silu(msg @ cW1[t]^T + cb1[t])  (reuses u1b)
    gemm(msgb, 512, wb_cW1 + (long)t*512*512, 512, cb1 + (long)t*512, nullptr,
         nullptr, u1b, 512, EEE, 512, 1);
    // w = hid @ cW2[t]^T ; scatter cd*w into sumv, count into cnt
    wtrans_kernel<<<EEE, 256, 0, stream>>>(u1b, cW2 + (long)t*4*512, cdt, rowI,
                                           sumv, cnt);
  }

  // ---- Phase 3: node update + coordinate update + edge logits ----
  // seg (f32, true columns) -> swizzled bf16 for the agg GEMM A operand
  conv(seg, 4096, segb, 4096, NN, 4096, 1);
  // agg = seg(N, T*H) @ relR(D, T*H)^T
  gemm(segb, 4096, wb_relR, 4096, nullptr, nullptr, agg, nullptr, 512, NN, 512, 0);
  // xcat = [h, agg] as swizzled bf16  (reuses msgb buffer)
  bf16_t* xcatb = msgb;
  xcat_kernel<<<((long)NN*1024 + 255) / 256, 256, 0, stream>>>(h, agg, xcatb,
                                                               (long)NN*1024);
  // tmp1 = silu(xcat @ nW1^T + nb1)  (reuses u1b)
  bf16_t* tmp1b = u1b;
  gemm(xcatb, 1024, wb_nW1, 1024, nb1, nullptr, nullptr, tmp1b, 512, NN, 512, 1);
  // h_new = h + tmp1 @ nW2^T + nb2 -> d_out (f32) AND swizzled bf16 (hb reuse)
  gemm(tmp1b, 512, wb_nW2, 512, nb2, h, out_h, hb, 512, NN, 512, 0);
  // x_new = coord + sumv / max(cnt, 1)
  xnew_kernel<<<((long)NN*12 + 255) / 256, 256, 0, stream>>>(coord, sumv, cnt,
                                                             out_x, (long)NN*12);
  // Qa/Qb = h_new @ eW1 halves  (reuse Pa/Pb)
  float* Qa = Pa; float* Qb = Pb;
  gemm(hb, 512, wb_eW1a, 512, nullptr, nullptr, Qa, nullptr, 512, NN, 512, 0);
  gemm(hb, 512, wb_eW1b, 512, nullptr, nullptr, Qb, nullptr, 512, NN, 512, 0);
  // m per edge
  m_kernel<<<TTT * EEE, 256, 0, stream>>>(Qa, Qb, edge_attr, edge_index,
                                          eW1, eb1, eW2, eb2, out_m);
}
